// SelfAttention_4973572128819
// MI455X (gfx1250) — matrix operand order repaired and checked by reading, not compiled, not run on hardware
//
#include <hip/hip_runtime.h>
#include <cstdint>

#define DEV __device__ __forceinline__

typedef __attribute__((ext_vector_type(16))) _Float16 v16h;
typedef __attribute__((ext_vector_type(8)))  float    v8f;
typedef __attribute__((ext_vector_type(4)))  int      v4i;
typedef __attribute__((ext_vector_type(4)))  unsigned int u32x4;
typedef __attribute__((ext_vector_type(8)))  int      i32x8;
typedef __attribute__((ext_vector_type(4)))  int      i32x4;

// ---------------- problem dims ----------------
constexpr int B    = 32;
constexpr int C    = 256;
constexpr int HW   = 4096;          // 64*64
constexpr int G    = 8;
constexpr int CPG  = C / G;         // 32 channels per group
constexpr int MQKV = 3 * C;         // 768
constexpr float GN_EPS     = 1e-5f;
constexpr float ATTN_SCALE = 0.0625f;   // 1/sqrt(256)

// ---------------- GEMM tiling ----------------
constexpr int KSTEP = 32;           // one WMMA K per LDS tile
constexpr int LDSS  = 40;           // LDS row stride in f16 (32 + 8 pad); 80B rows, 16B-aligned
constexpr int LBUF  = 128 * LDSS;   // one LDS buffer (f16 elements)

// ---------------- CDNA5 async global->LDS (ASYNCcnt) ----------------
DEV void async_ld_b128(void* lds, const void* g) {
#if __has_builtin(__builtin_amdgcn_global_load_async_to_lds_b128)
    __builtin_amdgcn_global_load_async_to_lds_b128(
        (__attribute__((address_space(1))) v4i*)g,
        (__attribute__((address_space(3))) v4i*)lds, 0, 0);
#else
    uint32_t loff = (uint32_t)(size_t)lds;
    asm volatile("global_load_async_to_lds_b128 %0, %1, off"
                 :: "v"(loff), "v"(g) : "memory");
#endif
}

#if __has_builtin(__builtin_amdgcn_s_wait_asynccnt)
#define WAIT_ASYNC(N) __builtin_amdgcn_s_wait_asynccnt(N)
#else
#define WAIT_ASYNC(N) asm volatile("s_wait_asynccnt %0" :: "n"(N) : "memory")
#endif

#if __has_builtin(__builtin_amdgcn_s_wait_tensorcnt)
#define WAIT_TENSOR(N) __builtin_amdgcn_s_wait_tensorcnt(N)
#else
#define WAIT_TENSOR(N) asm volatile("s_wait_tensorcnt %0" :: "n"(N) : "memory")
#endif

// ---------------- CDNA5 Tensor Data Mover (TENSORcnt) ----------------
#if __has_builtin(__builtin_amdgcn_tensor_load_to_lds)
#define HAVE_TDM 1
// 2-D tile load: 128 rows x 32 f16 (64B) per row, row pitch in LDS = 80B
// (pad_interval=3 -> pad after 16 DWORDs, pad_amount=3 -> 4 DWORDs of pad).
DEV void tdm_load_tile(uint32_t lds_byte, const void* g, int ld_elems) {
    const uint64_t ga = (uint64_t)(size_t)g;
    u32x4 g0 = {0u, 0u, 0u, 0u};
    g0[0] = 1u;                                   // count=1 valid descriptor
    g0[1] = lds_byte;                             // lds_addr
    g0[2] = (uint32_t)ga;                         // global_addr[31:0]
    g0[3] = (uint32_t)((ga >> 32) & 0x01FFFFFFu)  // global_addr[56:32]
          | (2u << 30);                           // type = 2 ("image")
    i32x8 g1 = {0, 0, 0, 0, 0, 0, 0, 0};
    g1[0] = (int)((1u << 16)                      // data_size = 2 bytes
                | (1u << 20)                      // pad_enable
                | (3u << 22)                      // pad_interval: 16 DWORDs (64B)
                | (3u << 25));                    // pad_amount:   4 DWORDs (16B)
    g1[1] = (int)(32u  << 16);                    // tensor_dim0 = 32
    g1[2] = (int)(128u << 16);                    // tensor_dim1 = 128
    g1[3] = (int)(32u  << 16);                    // tile_dim0 = 32
    g1[4] = 128;                                  // tile_dim1 = 128
    g1[5] = ld_elems;                             // tensor_dim0_stride
    i32x4 z4 = {0, 0, 0, 0};
#if defined(__clang_major__) && __clang_major__ >= 23
    i32x8 z8 = {0, 0, 0, 0, 0, 0, 0, 0};
    __builtin_amdgcn_tensor_load_to_lds(g0, g1, z4, z4, z8, 0);
#else
    __builtin_amdgcn_tensor_load_to_lds(g0, g1, z4, z4, 0);
#endif
}
#else
#define HAVE_TDM 0
#endif

// ------------- WMMA fragment loaders (wave32 CDNA5 layouts) -------------
DEV v16h frag_a(const _Float16* row, int klo) {
    union { v16h h; uint4 u[2]; } r;
    r.u[0] = *(const uint4*)(row + klo);
    r.u[1] = *(const uint4*)(row + klo + 8);
    return r.h;
}
DEV v16h frag_b(const _Float16* row, int kb) {
    union { v16h h; uint4 u[2]; } r;
    r.u[0] = *(const uint4*)(row + kb);
    r.u[1] = *(const uint4*)(row + kb + 8);
    return r.h;
}

// ------------- 128x128 block GEMM core, 256 threads = 8 waves -------------
// A  : streamed per-lane via global_load_async_to_lds_b128 (ASYNCcnt)
// Bt : streamed per-stage by the Tensor Data Mover, wave 0 issues (TENSORcnt)
// Both double-buffered; each wave owns a 32x64 tile -> c[2][4] accumulators.
DEV void gemm128(const _Float16* __restrict__ A, int lda,
                 const _Float16* __restrict__ Bt, int ldb, int K,
                 _Float16* As, _Float16* Bs,   // each 2*LBUF f16
                 v8f c[2][4]) {
    const int t    = threadIdx.x;
    const int lane = t & 31;
    const int wave = t >> 5;
    const int wm   = (wave & 3) * 32;
    const int wn   = (wave >> 2) * 64;
    const int klo  = (lane < 16) ? 0 : 16;
    const int kb   = (lane < 16) ? 0 : 16;
    const int fr   = lane & 15;
    const int cr   = t >> 1;            // copy row (0..127)
    const int co   = (t & 1) * 16;      // copy col offset (f16 units)

    const _Float16* ga = A + (size_t)cr * lda + co;
    _Float16* la = As + cr * LDSS + co;
#if HAVE_TDM
    const bool tdmw = (t < 32);                        // wave 0 drives the TDM
    const uint32_t bsb = (uint32_t)(size_t)Bs;         // LDS byte address of Bs
#else
    const _Float16* gbp = Bt + (size_t)cr * ldb + co;
    _Float16* lb = Bs + cr * LDSS + co;
#endif

    // pipeline prologue: stage 0 in flight
    async_ld_b128(la, ga);
#if HAVE_TDM
    if (tdmw) tdm_load_tile(bsb, Bt, ldb);
#else
    async_ld_b128(lb, gbp);
#endif

    const int S = K / KSTEP;
    for (int s = 0; s < S; ++s) {
        const int cur = (s & 1) * LBUF;
        if (s + 1 < S) {
            const int nxt = ((s + 1) & 1) * LBUF;
            async_ld_b128(la + nxt, ga + (size_t)(s + 1) * KSTEP);
#if HAVE_TDM
            if (tdmw) tdm_load_tile(bsb + (uint32_t)nxt * 2u,
                                    Bt + (size_t)(s + 1) * KSTEP, ldb);
            WAIT_ASYNC(1);                 // in-order: stage-s A has landed
            if (tdmw) WAIT_TENSOR(1);      // in-order: stage-s B tile has landed
#else
            async_ld_b128(lb + nxt, gbp + (size_t)(s + 1) * KSTEP);
            WAIT_ASYNC(2);
#endif
        } else {
            WAIT_ASYNC(0);
#if HAVE_TDM
            if (tdmw) WAIT_TENSOR(0);
#endif
        }
        __syncthreads();                   // all DMA for stage s visible to all waves

        v16h af[2], bf[4];
#pragma unroll
        for (int mi = 0; mi < 2; ++mi)
            af[mi] = frag_a(As + cur + (wm + mi * 16 + fr) * LDSS, klo);
#pragma unroll
        for (int ni = 0; ni < 4; ++ni)
            bf[ni] = frag_b(Bs + cur + (wn + ni * 16 + fr) * LDSS, kb);
#pragma unroll
        for (int mi = 0; mi < 2; ++mi)
#pragma unroll
            for (int ni = 0; ni < 4; ++ni)
                c[mi][ni] = __builtin_amdgcn_wmma_f32_16x16x32_f16(
                    false, af[mi], false, bf[ni], (short)0, c[mi][ni], false, false);
        __syncthreads();                   // done reading buf[cur] before refill
    }
}

DEV void zero_acc(v8f c[2][4]) {
#pragma unroll
    for (int i = 0; i < 2; ++i)
#pragma unroll
        for (int j = 0; j < 4; ++j)
#pragma unroll
            for (int e = 0; e < 8; ++e) c[i][j][e] = 0.0f;
}

// ================= Kernel 1: GroupNorm statistics =================
__global__ void __launch_bounds__(256) k_gnstats(const float* __restrict__ x,
                                                 float* __restrict__ mean,
                                                 float* __restrict__ rstd) {
    const int bg = blockIdx.x;                      // b*8 + g
    const float4* p = (const float4*)(x + (size_t)bg * CPG * HW);
    float s = 0.f, ss = 0.f;
    for (int i = threadIdx.x; i < CPG * HW / 4; i += 256) {
        float4 v = p[i];
        s  += v.x + v.y + v.z + v.w;
        ss += v.x * v.x + v.y * v.y + v.z * v.z + v.w * v.w;
    }
#pragma unroll
    for (int off = 16; off > 0; off >>= 1) {
        s  += __shfl_xor(s,  off, 32);
        ss += __shfl_xor(ss, off, 32);
    }
    __shared__ float red[16];
    const int wave = threadIdx.x >> 5, lane = threadIdx.x & 31;
    if (lane == 0) { red[wave] = s; red[wave + 8] = ss; }
    __syncthreads();
    if (threadIdx.x == 0) {
        float s1 = 0.f, s2 = 0.f;
#pragma unroll
        for (int w = 0; w < 8; ++w) { s1 += red[w]; s2 += red[w + 8]; }
        const float inv = 1.0f / (float)(CPG * HW);
        float m   = s1 * inv;
        float var = s2 * inv - m * m;
        mean[bg] = m;
        rstd[bg] = rsqrtf(var + GN_EPS);
    }
}

// ========== Kernel 2: GN apply + transpose to xnT [B][N][C] (f16) ==========
__global__ void __launch_bounds__(256) k_gnorm_t(const float* __restrict__ x,
                                                 const float* __restrict__ gamma,
                                                 const float* __restrict__ beta,
                                                 const float* __restrict__ mean,
                                                 const float* __restrict__ rstd,
                                                 _Float16* __restrict__ xnT) {
    __shared__ float tile[32][33];
    const int tn = blockIdx.x, tc = blockIdx.y, b = blockIdx.z;
    const int lane = threadIdx.x & 31, r0 = threadIdx.x >> 5;
#pragma unroll
    for (int i = 0; i < 4; ++i) {
        const int cl = r0 + 8 * i;
        const int cg = tc * 32 + cl;
        const int n  = tn * 32 + lane;
        const float m  = mean[b * G + (cg >> 5)];
        const float rs = rstd[b * G + (cg >> 5)];
        const float sc = gamma[cg] * rs;
        const float v  = x[((size_t)b * C + cg) * HW + n];
        tile[cl][lane] = (v - m) * sc + beta[cg];
    }
    __syncthreads();
#pragma unroll
    for (int i = 0; i < 4; ++i) {
        const int nl = r0 + 8 * i;
        xnT[((size_t)b * HW + tn * 32 + nl) * C + tc * 32 + lane] = (_Float16)tile[lane][nl];
    }
}

// ================= Kernel 3: pack weights f32 -> f16 =================
__global__ void __launch_bounds__(256) k_packw(const float* __restrict__ wqkv,
                                               const float* __restrict__ wproj,
                                               _Float16* __restrict__ Wq,
                                               _Float16* __restrict__ Wp) {
    const int i = blockIdx.x * 256 + threadIdx.x;       // 0 .. 262143
    if (i < MQKV * C) Wq[i] = (_Float16)wqkv[i];
    else              Wp[i - MQKV * C] = (_Float16)wproj[i - MQKV * C];
}

// ========== Kernel 4: QKV GEMM  qkv = Wq(768x256) x xn(256x4096) + bias ==========
// writes q [B][C][N] (pre-scaled by 1/16), k [B][C][N], vT [B][N][C]
__global__ void __launch_bounds__(256) k_qkv(const _Float16* __restrict__ Wq,
                                             const _Float16* __restrict__ xnT,
                                             const float* __restrict__ bias,
                                             _Float16* __restrict__ q,
                                             _Float16* __restrict__ kmat,
                                             _Float16* __restrict__ vT) {
    __shared__ _Float16 As[2 * LBUF], Bs[2 * LBUF];
    const int bn = blockIdx.x, bm = blockIdx.y, b = blockIdx.z;
    v8f c[2][4]; zero_acc(c);
    gemm128(Wq + (size_t)bm * 128 * C, C,
            xnT + ((size_t)b * HW + bn * 128) * C, C, C, As, Bs, c);

    const int t = threadIdx.x, lane = t & 31, wave = t >> 5;
    const int wm = (wave & 3) * 32, wn = (wave >> 2) * 64;
    const int msub = (lane < 16) ? 0 : 8, nf = lane & 15;
    const int seg  = bm >> 1;               // 0=q 1=k 2=v (M blocks are 128, segments 256)
    const int mseg = (bm & 1) * 128;
#pragma unroll
    for (int mi = 0; mi < 2; ++mi)
#pragma unroll
        for (int ni = 0; ni < 4; ++ni) {
            const int n  = bn * 128 + wn + ni * 16 + nf;
            const int mb = mseg + wm + mi * 16 + msub;  // row within segment
            if (seg == 2) {
                union { _Float16 h[8]; uint4 u; } o;
#pragma unroll
                for (int j = 0; j < 8; ++j)
                    o.h[j] = (_Float16)(c[mi][ni][j] + bias[512 + mb + j]);
                *(uint4*)(vT + ((size_t)b * HW + n) * C + mb) = o.u;
            } else {
                _Float16* dst = (seg == 0) ? q : kmat;
                const float sc = (seg == 0) ? ATTN_SCALE : 1.0f;
#pragma unroll
                for (int j = 0; j < 8; ++j) {
                    const int m = mb + j;
                    dst[((size_t)b * C + m) * HW + n] =
                        (_Float16)((c[mi][ni][j] + bias[seg * 256 + m]) * sc);
                }
            }
        }
}

// ========== Kernel 5: scores = q(256xN) x k(256xN)^T  (K = 4096) ==========
__global__ void __launch_bounds__(256) k_scores(const _Float16* __restrict__ q,
                                                const _Float16* __restrict__ kmat,
                                                float* __restrict__ S) {
    __shared__ _Float16 As[2 * LBUF], Bs[2 * LBUF];
    const int bn = blockIdx.x, bm = blockIdx.y, b = blockIdx.z;
    v8f c[2][4]; zero_acc(c);
    gemm128(q    + ((size_t)b * C + bm * 128) * HW, HW,
            kmat + ((size_t)b * C + bn * 128) * HW, HW, HW, As, Bs, c);

    const int t = threadIdx.x, lane = t & 31, wave = t >> 5;
    const int wm = (wave & 3) * 32, wn = (wave >> 2) * 64;
    const int msub = (lane < 16) ? 0 : 8, nf = lane & 15;
#pragma unroll
    for (int mi = 0; mi < 2; ++mi)
#pragma unroll
        for (int ni = 0; ni < 4; ++ni) {
            const int n = bn * 128 + wn + ni * 16 + nf;
            const int mb = bm * 128 + wm + mi * 16 + msub;
#pragma unroll
            for (int j = 0; j < 8; ++j)
                S[((size_t)b * C + mb + j) * C + n] = c[mi][ni][j];
        }
}

// ================= Kernel 6: softmax over rows of S (256 wide) =================
__global__ void __launch_bounds__(256) k_softmax(const float* __restrict__ S,
                                                 _Float16* __restrict__ attn) {
    const int row  = blockIdx.x * 8 + (threadIdx.x >> 5);   // one wave per row
    const int lane = threadIdx.x & 31;
    const float* sp = S + (size_t)row * C + lane * 8;
    float4 v0 = ((const float4*)sp)[0];
    float4 v1 = ((const float4*)sp)[1];
    float vals[8] = {v0.x, v0.y, v0.z, v0.w, v1.x, v1.y, v1.z, v1.w};
    float mx = vals[0];
#pragma unroll
    for (int j = 1; j < 8; ++j) mx = fmaxf(mx, vals[j]);
#pragma unroll
    for (int off = 16; off > 0; off >>= 1) mx = fmaxf(mx, __shfl_xor(mx, off, 32));
    float sum = 0.f;
#pragma unroll
    for (int j = 0; j < 8; ++j) { vals[j] = __expf(vals[j] - mx); sum += vals[j]; }
#pragma unroll
    for (int off = 16; off > 0; off >>= 1) sum += __shfl_xor(sum, off, 32);
    const float inv = 1.0f / sum;
    union { _Float16 h[8]; uint4 u; } o;
#pragma unroll
    for (int j = 0; j < 8; ++j) o.h[j] = (_Float16)(vals[j] * inv);
    *(uint4*)(attn + (size_t)row * C + lane * 8) = o.u;
}

// ========== Kernel 7: out = attn(256x256) x v(256xN) -> outT [B][N][C] ==========
__global__ void __launch_bounds__(256) k_attnv(const _Float16* __restrict__ attn,
                                               const _Float16* __restrict__ vT,
                                               _Float16* __restrict__ outT) {
    __shared__ _Float16 As[2 * LBUF], Bs[2 * LBUF];
    const int bn = blockIdx.x, bm = blockIdx.y, b = blockIdx.z;
    v8f c[2][4]; zero_acc(c);
    gemm128(attn + (size_t)b * C * C + bm * 128 * C, C,
            vT   + ((size_t)b * HW + bn * 128) * C, C, C, As, Bs, c);

    const int t = threadIdx.x, lane = t & 31, wave = t >> 5;
    const int wm = (wave & 3) * 32, wn = (wave >> 2) * 64;
    const int msub = (lane < 16) ? 0 : 8, nf = lane & 15;
#pragma unroll
    for (int mi = 0; mi < 2; ++mi)
#pragma unroll
        for (int ni = 0; ni < 4; ++ni) {
            const int n  = bn * 128 + wn + ni * 16 + nf;
            const int mb = bm * 128 + wm + mi * 16 + msub;
            union { _Float16 h[8]; uint4 u; } o;
#pragma unroll
            for (int j = 0; j < 8; ++j) o.h[j] = (_Float16)c[mi][ni][j];
            *(uint4*)(outT + ((size_t)b * HW + n) * C + mb) = o.u;
        }
}

// ========== Kernel 8: final = x + Wp(256x256) x out(256xN) + bias ==========
__global__ void __launch_bounds__(256) k_proj(const _Float16* __restrict__ Wp,
                                              const _Float16* __restrict__ outT,
                                              const float* __restrict__ pbias,
                                              const float* __restrict__ x,
                                              float* __restrict__ out) {
    __shared__ _Float16 As[2 * LBUF], Bs[2 * LBUF];
    const int bn = blockIdx.x, bm = blockIdx.y, b = blockIdx.z;
    v8f c[2][4]; zero_acc(c);
    gemm128(Wp   + (size_t)bm * 128 * C, C,
            outT + ((size_t)b * HW + bn * 128) * C, C, C, As, Bs, c);

    const int t = threadIdx.x, lane = t & 31, wave = t >> 5;
    const int wm = (wave & 3) * 32, wn = (wave >> 2) * 64;
    const int msub = (lane < 16) ? 0 : 8, nf = lane & 15;
#pragma unroll
    for (int mi = 0; mi < 2; ++mi)
#pragma unroll
        for (int ni = 0; ni < 4; ++ni) {
            const int n  = bn * 128 + wn + ni * 16 + nf;
            const int mb = bm * 128 + wm + mi * 16 + msub;
#pragma unroll
            for (int j = 0; j < 8; ++j) {
                const int o = mb + j;
                const size_t idx = ((size_t)b * C + o) * HW + n;
                out[idx] = x[idx] + c[mi][ni][j] + pbias[o];
            }
        }
}

// ---------------- workspace layout ----------------
constexpr size_t SZ_MAT   = sizeof(_Float16) * (size_t)B * C * HW;   // 64 MiB
constexpr size_t OFF_MEAN = 0;
constexpr size_t OFF_RSTD = 1024;
constexpr size_t OFF_WQ   = 2048;
constexpr size_t OFF_WP   = OFF_WQ + sizeof(_Float16) * (size_t)MQKV * C;
constexpr size_t OFF_XNT  = (OFF_WP + sizeof(_Float16) * (size_t)C * C + 255) & ~(size_t)255;
constexpr size_t OFF_Q    = OFF_XNT + SZ_MAT;
constexpr size_t OFF_K    = OFF_Q + SZ_MAT;
constexpr size_t OFF_VT   = OFF_K + SZ_MAT;
constexpr size_t OFF_S    = OFF_VT + SZ_MAT;
constexpr size_t OFF_ATTN = OFF_S + sizeof(float) * (size_t)B * C * C;
constexpr size_t OFF_OUTT = OFF_XNT;   // xnT is dead after k_qkv -> reuse for outT

extern "C" void kernel_launch(void* const* d_in, const int* in_sizes, int n_in,
                              void* d_out, int out_size, void* d_ws, size_t ws_size,
                              hipStream_t stream) {
    (void)in_sizes; (void)n_in; (void)out_size; (void)ws_size;
    const float* x    = (const float*)d_in[0];
    const float* gw   = (const float*)d_in[1];
    const float* gb   = (const float*)d_in[2];
    const float* wqkv = (const float*)d_in[3];
    const float* bqkv = (const float*)d_in[4];
    const float* wpro = (const float*)d_in[5];
    const float* bpro = (const float*)d_in[6];
    float* out = (float*)d_out;

    char* ws = (char*)d_ws;
    float*    mean = (float*)(ws + OFF_MEAN);
    float*    rstd = (float*)(ws + OFF_RSTD);
    _Float16* Wq   = (_Float16*)(ws + OFF_WQ);
    _Float16* Wp   = (_Float16*)(ws + OFF_WP);
    _Float16* xnT  = (_Float16*)(ws + OFF_XNT);
    _Float16* q    = (_Float16*)(ws + OFF_Q);
    _Float16* kmat = (_Float16*)(ws + OFF_K);
    _Float16* vT   = (_Float16*)(ws + OFF_VT);
    float*    S    = (float*)(ws + OFF_S);
    _Float16* attn = (_Float16*)(ws + OFF_ATTN);
    _Float16* outT = (_Float16*)(ws + OFF_OUTT);

    k_gnstats <<<B * G, 256, 0, stream>>>(x, mean, rstd);
    k_gnorm_t <<<dim3(HW / 32, C / 32, B), 256, 0, stream>>>(x, gw, gb, mean, rstd, xnT);
    k_packw   <<<(MQKV * C + C * C) / 256, 256, 0, stream>>>(wqkv, wpro, Wq, Wp);
    k_qkv     <<<dim3(HW / 128, MQKV / 128, B), 256, 0, stream>>>(Wq, xnT, bqkv, q, kmat, vT);
    k_scores  <<<dim3(C / 128, C / 128, B), 256, 0, stream>>>(q, kmat, S);
    k_softmax <<<B * C / 8, 256, 0, stream>>>(S, attn);
    k_attnv   <<<dim3(HW / 128, C / 128, B), 256, 0, stream>>>(attn, vT, outT);
    k_proj    <<<dim3(HW / 128, C / 128, B), 256, 0, stream>>>(Wp, outT, bpro, x, out);
}
